// Mapper_49074296324497
// MI455X (gfx1250) — matrix operand order repaired and checked by reading, not compiled, not run on hardware
//
#include <hip/hip_runtime.h>

// Problem constants (from reference)
#define SEQ  512
#define BZ   64
#define DIN  1024
#define HID  256
#define DOUT 1024
#define NEXP 16

// Tiling
#define MT   64      // seq rows per workgroup
#define KCH  64      // K chunk staged per iteration

// LDS pitches (elements, 16-byte-multiple rows)
#define A_PITCH 72
#define W_PITCH 72
#define H_PITCH 264

typedef __attribute__((ext_vector_type(16))) __bf16 bf16x16;
typedef __attribute__((ext_vector_type(8)))  __bf16 bf16x8;
typedef __attribute__((ext_vector_type(8)))  float  f32x8;

__device__ __forceinline__ unsigned short f2bf(float f) {
  // round-to-nearest-even f32 -> bf16
  unsigned int u = __float_as_uint(f);
  u += 0x7FFFu + ((u >> 16) & 1u);
  return (unsigned short)(u >> 16);
}

__device__ __forceinline__ unsigned int pk2(float a, float b) {
  return (unsigned)f2bf(a) | ((unsigned)f2bf(b) << 16);
}

__device__ __forceinline__ bf16x8 ld8(const unsigned short* p) {
  return *reinterpret_cast<const bf16x8*>(p);
}

// Build a v16 fragment from two contiguous 8-element runs (ds_load_b128 x2)
__device__ __forceinline__ bf16x16 frag2(const unsigned short* p0,
                                         const unsigned short* p1) {
  bf16x8 lo = ld8(p0);
  bf16x8 hi = ld8(p1);
  return __builtin_shufflevector(lo, hi, 0, 1, 2, 3, 4, 5, 6, 7,
                                 8, 9, 10, 11, 12, 13, 14, 15);
}

__device__ __forceinline__ f32x8 wmma_bf16(bf16x16 a, bf16x16 b, f32x8 c) {
  return __builtin_amdgcn_wmma_f32_16x16x32_bf16(
      false, a, false, b, (short)0, c, false, false);
}

// ---------------------------------------------------------------------------
// Prep kernel: W[e][K][N] f32  ->  Wt[e][N][K] bf16   (convert + transpose)
// grid: (N/64, K/64, NEXP), block 256. Tile transpose through LDS so both
// global reads and global writes are wide/coalesced.
// ---------------------------------------------------------------------------
__global__ __launch_bounds__(256) void cvt_transpose_kernel(
    const float* __restrict__ W, unsigned short* __restrict__ Wt,
    int K, int N) {
  __shared__ __align__(16) unsigned short tile[64 * A_PITCH];  // [n][k], pitch 72

  const int e  = blockIdx.z;
  const int k0 = blockIdx.y * 64;
  const int n0 = blockIdx.x * 64;
  const float* Wp = W + ((size_t)e * K + k0) * N + n0;
  unsigned short* Wo = Wt + ((size_t)e * N + n0) * K + k0;
  const int t = threadIdx.x;

#pragma unroll
  for (int i = 0; i < 16; ++i) {
    const int flat = t + i * 256;      // 0..4095
    const int k = flat >> 6;           // 64 rows of W
    const int n = flat & 63;           // coalesced along n
    tile[n * A_PITCH + k] = f2bf(Wp[(size_t)k * N + n]);
  }
  __syncthreads();
#pragma unroll
  for (int i = 0; i < 2; ++i) {
    const int flat = t + i * 256;      // 512 groups of 8 bf16
    const int n = flat >> 3;
    const int g = flat & 7;
    uint4 v = *reinterpret_cast<const uint4*>(&tile[n * A_PITCH + g * 8]);
    *reinterpret_cast<uint4*>(&Wo[(size_t)n * K + g * 8]) = v;
  }
}

// ---------------------------------------------------------------------------
// Main fused MoE MLP kernel. 256 threads = 8 waves; MT=64 seq rows per block.
// Each wave owns 8 output tiles: row-block mt = wave&3, col group (wave>>2)*8.
// ---------------------------------------------------------------------------
__global__ __launch_bounds__(256) void moe_mlp_kernel(
    const float* __restrict__ x, const int* __restrict__ lang,
    const unsigned short* __restrict__ W1t,  // [E][HID][DIN] bf16
    const float* __restrict__ b1,
    const unsigned short* __restrict__ W2t,  // [E][DOUT][HID] bf16
    const float* __restrict__ b2, float* __restrict__ y) {
  __shared__ __align__(16) unsigned short a_s[MT * A_PITCH];   // x tile, bf16
  __shared__ __align__(16) unsigned short w_s[256 * W_PITCH];  // W chunk [n][k]
  __shared__ __align__(16) unsigned short h_s[MT * H_PITCH];   // hidden acts

  const int t    = threadIdx.x;
  const int lane = t & 31;
  const int wave = t >> 5;
  const int ln   = lane & 15;        // N (or M) index within tile
  const int hl   = lane >> 4;        // lane-half selects K run / M half
  const int mt   = wave & 3;         // row block (16 rows each, 4 blocks)
  const int ntg  = (wave >> 2) * 8;  // this wave's 8 column tiles
  const int b    = blockIdx.y;
  const int s0   = blockIdx.x * MT;
  const int e    = lang[b];

  const unsigned short* W1g = W1t + (size_t)e * HID * DIN;
  const float*          b1g = b1 + (size_t)e * HID;
  const unsigned short* W2g = W2t + (size_t)e * DOUT * HID;
  const float*          b2g = b2 + (size_t)e * DOUT;

  f32x8 acc[8];
#pragma unroll
  for (int i = 0; i < 8; ++i)
    acc[i] = f32x8{0.f, 0.f, 0.f, 0.f, 0.f, 0.f, 0.f, 0.f};

  // ---------------- GEMM1: h = relu(x @ W1 + b1), K = DIN ----------------
  for (int kc = 0; kc < DIN; kc += KCH) {
    {  // stage x tile [MT x KCH] f32 -> bf16 (16 floats/thread)
      const int r = t >> 2, g = t & 3;
      const float* xp = x + ((size_t)(s0 + r) * BZ + b) * DIN + kc + g * 16;
      float4 v0 = *reinterpret_cast<const float4*>(xp);
      float4 v1 = *reinterpret_cast<const float4*>(xp + 4);
      float4 v2 = *reinterpret_cast<const float4*>(xp + 8);
      float4 v3 = *reinterpret_cast<const float4*>(xp + 12);
      uint4 p0, p1;
      p0.x = pk2(v0.x, v0.y); p0.y = pk2(v0.z, v0.w);
      p0.z = pk2(v1.x, v1.y); p0.w = pk2(v1.z, v1.w);
      p1.x = pk2(v2.x, v2.y); p1.y = pk2(v2.z, v2.w);
      p1.z = pk2(v3.x, v3.y); p1.w = pk2(v3.z, v3.w);
      unsigned short* dst = &a_s[r * A_PITCH + g * 16];
      *reinterpret_cast<uint4*>(dst) = p0;
      *reinterpret_cast<uint4*>(dst + 8) = p1;
    }
    {  // stage W1 chunk: pure wide copy, already bf16 + transposed
      const unsigned short* wp = W1g + kc;  // row n stride DIN
#pragma unroll
      for (int i = 0; i < 8; ++i) {
        const int flat = t + i * 256;  // 2048 groups of 8 bf16
        const int n = flat >> 3;
        const int g = flat & 7;
        uint4 v = *reinterpret_cast<const uint4*>(wp + (size_t)n * DIN + g * 8);
        *reinterpret_cast<uint4*>(&w_s[n * W_PITCH + g * 8]) = v;
      }
    }
    __syncthreads();
#pragma unroll
    for (int ks = 0; ks < 2; ++ks) {
      const int kb = ks * 32;
      const unsigned short* arow = &a_s[(mt * 16 + ln) * A_PITCH];
      bf16x16 af = frag2(arow + kb + 8 * hl, arow + kb + 16 + 8 * hl);
#pragma unroll
      for (int i = 0; i < 8; ++i) {
        const unsigned short* brow = &w_s[((ntg + i) * 16 + ln) * W_PITCH];
        bf16x16 bfr = frag2(brow + kb + 8 * hl, brow + kb + 16 + 8 * hl);
        acc[i] = wmma_bf16(af, bfr, acc[i]);
      }
    }
    __syncthreads();
  }

  // epilogue 1: bias + relu, store h as bf16 in LDS
#pragma unroll
  for (int i = 0; i < 8; ++i) {
    const int n = (ntg + i) * 16 + ln;
    const float bias = b1g[n];
#pragma unroll
    for (int r = 0; r < 8; ++r) {
      const int m = mt * 16 + hl * 8 + r;
      float v = acc[i][r] + bias;
      h_s[m * H_PITCH + n] = f2bf(v > 0.f ? v : 0.f);
    }
  }
  __syncthreads();

  // ---------------- GEMM2: y = h @ W2 + b2, K = HID, N streamed ----------------
  for (int nc = 0; nc < DOUT; nc += 256) {
#pragma unroll
    for (int i = 0; i < 8; ++i)
      acc[i] = f32x8{0.f, 0.f, 0.f, 0.f, 0.f, 0.f, 0.f, 0.f};

    for (int kc = 0; kc < HID; kc += KCH) {
      {  // stage W2 chunk: wide copy
        const unsigned short* wp = W2g + (size_t)nc * HID + kc;  // rows stride HID
#pragma unroll
        for (int i = 0; i < 8; ++i) {
          const int flat = t + i * 256;
          const int n = flat >> 3;
          const int g = flat & 7;
          uint4 v = *reinterpret_cast<const uint4*>(wp + (size_t)n * HID + g * 8);
          *reinterpret_cast<uint4*>(&w_s[n * W_PITCH + g * 8]) = v;
        }
      }
      __syncthreads();
#pragma unroll
      for (int ks = 0; ks < 2; ++ks) {
        const int kb = ks * 32;
        const unsigned short* arow = &h_s[(mt * 16 + ln) * H_PITCH + kc];
        bf16x16 af = frag2(arow + kb + 8 * hl, arow + kb + 16 + 8 * hl);
#pragma unroll
        for (int i = 0; i < 8; ++i) {
          const unsigned short* brow = &w_s[((ntg + i) * 16 + ln) * W_PITCH];
          bf16x16 bfr = frag2(brow + kb + 8 * hl, brow + kb + 16 + 8 * hl);
          acc[i] = wmma_bf16(af, bfr, acc[i]);
        }
      }
      __syncthreads();
    }

    // epilogue 2: bias, store y (f32)
#pragma unroll
    for (int i = 0; i < 8; ++i) {
      const int n = nc + (ntg + i) * 16 + ln;
      const float bias = b2g[n];
#pragma unroll
      for (int r = 0; r < 8; ++r) {
        const int m = mt * 16 + hl * 8 + r;
        y[((size_t)(s0 + m) * BZ + b) * DOUT + n] = acc[i][r] + bias;
      }
    }
  }
}

extern "C" void kernel_launch(void* const* d_in, const int* in_sizes, int n_in,
                              void* d_out, int out_size, void* d_ws,
                              size_t ws_size, hipStream_t stream) {
  const float* x  = (const float*)d_in[0];
  const int*   lg = (const int*)d_in[1];
  const float* W1 = (const float*)d_in[2];
  const float* b1 = (const float*)d_in[3];
  const float* W2 = (const float*)d_in[4];
  const float* b2 = (const float*)d_in[5];
  float* y = (float*)d_out;
  (void)in_sizes; (void)n_in; (void)out_size; (void)ws_size;

  // Workspace layout: Wt1 [16][256][1024] bf16 (8 MB), Wt2 [16][1024][256] bf16 (8 MB)
  unsigned short* W1t = (unsigned short*)d_ws;
  unsigned short* W2t = W1t + (size_t)NEXP * HID * DIN;

  // Pre-convert + pre-transpose weights (runs every call; deterministic).
  cvt_transpose_kernel<<<dim3(HID / 64, DIN / 64, NEXP), dim3(256), 0, stream>>>(
      W1, W1t, DIN, HID);
  cvt_transpose_kernel<<<dim3(DOUT / 64, HID / 64, NEXP), dim3(256), 0, stream>>>(
      W2, W2t, HID, DOUT);

  dim3 grid(SEQ / MT, BZ);
  moe_mlp_kernel<<<grid, dim3(256), 0, stream>>>(x, lg, W1t, b1, W2t, b2, y);
}
